// GraphAllEdgeNet_81381040324720
// MI455X (gfx1250) — matrix operand order repaired and checked by reading, not compiled, not run on hardware
//
#include <hip/hip_runtime.h>
#include <hip/hip_bf16.h>
#include <stdint.h>

// ---------------------------------------------------------------------------
// Types for CDNA5 WMMA (wave32): v_wmma_f32_16x16x32_bf16
// ---------------------------------------------------------------------------
typedef __attribute__((ext_vector_type(16))) __bf16 v16bf;
typedef __attribute__((ext_vector_type(8)))  float  v8f;

#define TILE_E 64       // edges per block (4 waves x 16 edges)
#define LDS_PITCH 264   // 256 features + 8 pad (132 dwords == 4 mod 64 banks)

__device__ __forceinline__ unsigned short f2bf(float f) {
  unsigned u = __float_as_uint(f);
  u += 0x7FFFu + ((u >> 16) & 1u);          // round-to-nearest-even
  return (unsigned short)(u >> 16);
}

// A fragment (16x32 bf16, MxK): lane L holds row M=L%16.
// elems 0..7  -> K = kbase + (L/16)*16 .. +7
// elems 8..15 -> K = kbase + (L/16)*16 + 8 .. +15
__device__ __forceinline__ v16bf load_a_frag(const unsigned short* row, int kbase, int hi) {
  v16bf a;
  const float4* p0 = reinterpret_cast<const float4*>(row + kbase + hi * 16);
  const float4* p1 = reinterpret_cast<const float4*>(row + kbase + hi * 16 + 8);
  float4* d = reinterpret_cast<float4*>(&a);
  d[0] = *p0;
  d[1] = *p1;
  return a;
}

// B fragment (32x16 bf16, KxN): lane L holds col N=L%16,
// elems 0..15 -> K = kbase + (L/16)*16 .. +15 (contiguous 32 bytes)
__device__ __forceinline__ v16bf load_b_frag(const unsigned short* p) {
  v16bf b;
  const float4* q = reinterpret_cast<const float4*>(p);
  float4* d = reinterpret_cast<float4*>(&b);
  d[0] = q[0];
  d[1] = q[1];
  return b;
}

// ---------------------------------------------------------------------------
// Edge-conv main kernel.
// MODE 0: compute t = relu(bn1(h)) @ w1^T, accumulate per-feature sum/sumsq
// MODE 1: recompute t, apply bn2+relu, GEMM with w2, atomic segment-max at dst
// ---------------------------------------------------------------------------
template <int MODE>
__global__ __launch_bounds__(128)
void edge_pass(const float* __restrict__ g, const long long* __restrict__ ei, long long E,
               const float* __restrict__ sc1, const float* __restrict__ sh1,
               const unsigned short* __restrict__ w1,
               const float* __restrict__ sc2v, const float* __restrict__ sh2v,
               const unsigned short* __restrict__ w2,
               float* __restrict__ sum2, float* __restrict__ sq2,
               unsigned int* __restrict__ nmax)
{
  __shared__ __align__(16) unsigned short sh_h[TILE_E][LDS_PITCH];
  __shared__ int sh_src[TILE_E];
  __shared__ int sh_dst[TILE_E];

  const int t = threadIdx.x;                       // 0..127
  const long long ebase = (long long)blockIdx.x * TILE_E;

  if (t < TILE_E) {
    long long e = ebase + t;
    int s = -1, d = -1;
    if (e < E) { s = (int)ei[e]; d = (int)ei[E + e]; }
    sh_src[t] = s;
    sh_dst[t] = d;
  }
  __syncthreads();

  // Stage h = bn1+relu of cat([xi, xj-xi]) as bf16 into LDS. Thread t owns
  // feature pair (t, t+128); gathers of g are L2-resident (g fits in 192MB L2).
  {
    const float a1 = sc1[t], b1 = sh1[t], a2 = sc1[t + 128], b2 = sh1[t + 128];
    for (int e = 0; e < TILE_E; ++e) {
      int d = sh_dst[e], s = sh_src[e];
      float h0 = 0.f, h1 = 0.f;
      if (d >= 0) {
        float xi = g[(long long)d * 128 + t];
        float xj = g[(long long)s * 128 + t];
        h0 = fmaxf(xi * a1 + b1, 0.f);
        h1 = fmaxf((xj - xi) * a2 + b2, 0.f);
      }
      sh_h[e][t]       = f2bf(h0);
      sh_h[e][t + 128] = f2bf(h1);
    }
  }
  __syncthreads();

  const int lane = t & 31;
  const int wv   = t >> 5;          // wave id 0..3
  const int lo   = lane & 15;
  const int hi   = lane >> 4;
  const int row0 = wv * 16;         // this wave's 16 edge rows

  const unsigned short* arow = &sh_h[row0 + lo][0];

  // GEMM1: [16 x 256] x [256 x 128] -> 8 accumulator tiles
  v8f acc[8] = {};
  #pragma unroll
  for (int kk = 0; kk < 8; ++kk) {
    v16bf af = load_a_frag(arow, kk * 32, hi);
    #pragma unroll
    for (int n = 0; n < 8; ++n) {
      v16bf bf = load_b_frag(w1 + ((n * 16 + lo) * 256 + kk * 32 + hi * 16));
      acc[n] = __builtin_amdgcn_wmma_f32_16x16x32_bf16(false, af, false, bf,
                                                       (short)0, acc[n], false, false);
    }
  }

  if (MODE == 0) {
    // bn2 statistics: reduce over this wave's 16 edge rows, atomics to global.
    #pragma unroll
    for (int n = 0; n < 8; ++n) {
      float s = 0.f, q = 0.f;
      #pragma unroll
      for (int r = 0; r < 8; ++r) { float v = acc[n][r]; s += v; q += v * v; }
      s += __shfl_xor(s, 16, 32);   // combine lane L with L^16 (same N col)
      q += __shfl_xor(q, 16, 32);
      if (hi == 0) {
        atomicAdd(&sum2[n * 16 + lo], s);
        atomicAdd(&sq2[n * 16 + lo], q);
      }
    }
  } else {
    // bn2 + relu, write t-tile back into LDS (cols 0..127 of own rows)
    #pragma unroll
    for (int n = 0; n < 8; ++n) {
      const int c = n * 16 + lo;
      const float sc = sc2v[c], sh = sh2v[c];
      #pragma unroll
      for (int r = 0; r < 8; ++r) {
        float v = fmaxf(acc[n][r] * sc + sh, 0.f);
        sh_h[row0 + hi * 8 + r][c] = f2bf(v);   // D layout: row=r+8*hi, col=lo
      }
    }
    __syncthreads();

    // GEMM2: [16 x 128] x [128 x 128]
    v8f acc2[8] = {};
    #pragma unroll
    for (int kk = 0; kk < 4; ++kk) {
      v16bf af = load_a_frag(arow, kk * 32, hi);
      #pragma unroll
      for (int n = 0; n < 8; ++n) {
        v16bf bf = load_b_frag(w2 + ((n * 16 + lo) * 128 + kk * 32 + hi * 16));
        acc2[n] = __builtin_amdgcn_wmma_f32_16x16x32_bf16(false, af, false, bf,
                                                          (short)0, acc2[n], false, false);
      }
    }

    // segment-max at dst via order-preserving uint atomicMax
    #pragma unroll
    for (int n = 0; n < 8; ++n) {
      const int c = n * 16 + lo;
      #pragma unroll
      for (int r = 0; r < 8; ++r) {
        int rr = row0 + hi * 8 + r;
        int node = sh_dst[rr];
        if (node >= 0) {
          unsigned u = __float_as_uint(acc2[n][r]);
          u = (u & 0x80000000u) ? ~u : (u | 0x80000000u);
          atomicMax(&nmax[(long long)node * 128 + c], u);
        }
      }
    }
  }
}

// ---------------------------------------------------------------------------
// Supporting kernels
// ---------------------------------------------------------------------------
__global__ void cvt_bf16_k(const float* __restrict__ s, unsigned short* __restrict__ d, int n) {
  for (int i = blockIdx.x * blockDim.x + threadIdx.x; i < n; i += gridDim.x * blockDim.x)
    d[i] = f2bf(s[i]);
}

// g = cat(x[:,0,:],x[:,1,:]) @ fusion_w^T + fusion_b   (x[n] is already the concat)
__global__ __launch_bounds__(128)
void fusion_k(const float* __restrict__ x, const float* __restrict__ w,
              const float* __restrict__ b, float* __restrict__ g, int N) {
  __shared__ float xs[256];
  int n = blockIdx.x;
  if (n >= N) return;
  int t = threadIdx.x;
  xs[t]       = x[(long long)n * 256 + t];
  xs[t + 128] = x[(long long)n * 256 + 128 + t];
  __syncthreads();
  float acc = b[t];
  const float* wr = w + t * 256;
  #pragma unroll 8
  for (int k = 0; k < 256; ++k) acc += xs[k] * wr[k];
  g[(long long)n * 128 + t] = acc;
}

// bn1 statistics over edges of h = cat([xi, xj-xi])
__global__ __launch_bounds__(128)
void bn1_stats_k(const float* __restrict__ g, const long long* __restrict__ ei, long long E,
                 float* __restrict__ sum, float* __restrict__ sq) {
  const int t = threadIdx.x;
  const long long base = (long long)blockIdx.x * 512;
  float s0 = 0.f, q0 = 0.f, s1 = 0.f, q1 = 0.f;
  for (int i = 0; i < 512; ++i) {
    long long e = base + i;
    if (e < E) {
      int sN = (int)ei[e];
      int dN = (int)ei[E + e];
      float a = g[(long long)dN * 128 + t];
      float b = g[(long long)sN * 128 + t];
      float h0 = a, h1 = b - a;
      s0 += h0; q0 += h0 * h0;
      s1 += h1; q1 += h1 * h1;
    }
  }
  atomicAdd(&sum[t], s0);        atomicAdd(&sq[t], q0);
  atomicAdd(&sum[128 + t], s1);  atomicAdd(&sq[128 + t], q1);
}

// scale = gamma/sqrt(var+eps), shift = beta - mean*scale  (biased variance)
__global__ void finalize_bn_k(const float* __restrict__ sum, const float* __restrict__ sq,
                              float inv_cnt, const float* __restrict__ gamma,
                              const float* __restrict__ beta,
                              float* __restrict__ scale, float* __restrict__ shift, int n) {
  int i = blockIdx.x * blockDim.x + threadIdx.x;
  if (i < n) {
    float m = sum[i] * inv_cnt;
    float v = fmaxf(sq[i] * inv_cnt - m * m, 0.f);
    float sc = gamma[i] * rsqrtf(v + 1e-5f);
    scale[i] = sc;
    shift[i] = beta[i] - m * sc;
  }
}

// decode ordered-uint max, empty/non-finite -> 0, optional residual add
__global__ void node_finalize_k(const unsigned* __restrict__ nmax, const float* __restrict__ gprev,
                                float* __restrict__ h, long long total, int add_res) {
  long long i = (long long)blockIdx.x * blockDim.x + threadIdx.x;
  if (i >= total) return;
  unsigned u = nmax[i];
  float f = 0.f;
  if (u != 0u) {
    unsigned bits = (u & 0x80000000u) ? (u & 0x7FFFFFFFu) : ~u;
    f = __uint_as_float(bits);
    if (!isfinite(f)) f = 0.f;
  }
  if (add_res) f += gprev[i];
  h[i] = f;
}

__global__ __launch_bounds__(128)
void node_stats_k(const float* __restrict__ h, int N, float* __restrict__ sum, float* __restrict__ sq) {
  const int t = threadIdx.x;
  const int base = blockIdx.x * 256;
  float s = 0.f, q = 0.f;
  for (int i = 0; i < 256; ++i) {
    int n = base + i;
    if (n < N) { float v = h[(long long)n * 128 + t]; s += v; q += v * v; }
  }
  atomicAdd(&sum[t], s);
  atomicAdd(&sq[t], q);
}

__global__ void apply_bn_relu_k(const float* __restrict__ h, const float* __restrict__ sc,
                                const float* __restrict__ sh, float* __restrict__ g, long long total) {
  long long i = (long long)blockIdx.x * blockDim.x + threadIdx.x;
  if (i >= total) return;
  int c = (int)(i & 127);
  g[i] = fmaxf(h[i] * sc[c] + sh[c], 0.f);
}

__global__ void final_linear_k(const float* __restrict__ h, const float* __restrict__ w,
                               const float* __restrict__ b, float* __restrict__ out, int N) {
  int n = blockIdx.x * blockDim.x + threadIdx.x;
  if (n >= N) return;
  float a0 = b[0], a1 = b[1];
  const float* hr = h + (long long)n * 128;
  #pragma unroll 8
  for (int c = 0; c < 128; ++c) {
    float v = hr[c];
    a0 += v * w[c];
    a1 += v * w[128 + c];
  }
  out[(long long)n * 2]     = a0;
  out[(long long)n * 2 + 1] = a1;
}

// ---------------------------------------------------------------------------
// Host orchestration (all on `stream`, graph-capture safe)
// ---------------------------------------------------------------------------
extern "C" void kernel_launch(void* const* d_in, const int* in_sizes, int n_in,
                              void* d_out, int out_size, void* d_ws, size_t ws_size,
                              hipStream_t stream) {
  const float*     x        = (const float*)d_in[0];
  const long long* ei       = (const long long*)d_in[1];
  const float*     fusion_w = (const float*)d_in[2];
  const float*     fusion_b = (const float*)d_in[3];
  const float*     bn1_g    = (const float*)d_in[4];
  const float*     bn1_b    = (const float*)d_in[5];
  const float*     fc1_w    = (const float*)d_in[6];
  const float*     bn2_g    = (const float*)d_in[7];
  const float*     bn2_b    = (const float*)d_in[8];
  const float*     fc2_w    = (const float*)d_in[9];
  const float*     bn_g     = (const float*)d_in[10];
  const float*     bn_b     = (const float*)d_in[11];
  const float*     fc_w     = (const float*)d_in[12];
  const float*     fc_b     = (const float*)d_in[13];
  float*           out      = (float*)d_out;

  const long long N = in_sizes[0] / 256;
  const long long E = in_sizes[1] / 2;

  char* ws = (char*)d_ws;
  size_t off = 0;
  auto take = [&](size_t bytes) -> void* {
    off = (off + 255) & ~(size_t)255;
    void* p = ws + off;
    off += bytes;
    return p;
  };
  float*          g    = (float*)take((size_t)N * 128 * 4);
  float*          h    = (float*)take((size_t)N * 128 * 4);
  unsigned*       nmax = (unsigned*)take((size_t)N * 128 * 4);
  unsigned short* w1b  = (unsigned short*)take((size_t)4 * 128 * 256 * 2);
  unsigned short* w2b  = (unsigned short*)take((size_t)4 * 128 * 128 * 2);
  float* sums  = (float*)take(1024 * 4);
  float* sum1 = sums, *sq1 = sums + 256, *sum2 = sums + 512, *sq2 = sums + 640,
       * sumn = sums + 768, *sqn = sums + 896;
  float* coefs = (float*)take(1024 * 4);
  float* sc1 = coefs, *sh1 = coefs + 256, *sc2 = coefs + 512, *sh2 = coefs + 640,
       * scn = coefs + 768, *shn = coefs + 896;

  cvt_bf16_k<<<256, 256, 0, stream>>>(fc1_w, w1b, 4 * 128 * 256);
  cvt_bf16_k<<<128, 256, 0, stream>>>(fc2_w, w2b, 4 * 128 * 128);
  fusion_k<<<(int)N, 128, 0, stream>>>(x, fusion_w, fusion_b, g, (int)N);

  const int gridE     = (int)((E + TILE_E - 1) / TILE_E);
  const int gridElems = (int)((N * 128 + 255) / 256);

  for (int l = 0; l < 4; ++l) {
    hipMemsetAsync(sums, 0, 1024 * 4, stream);
    hipMemsetAsync(nmax, 0, (size_t)N * 128 * 4, stream);

    bn1_stats_k<<<(int)((E + 511) / 512), 128, 0, stream>>>(g, ei, E, sum1, sq1);
    finalize_bn_k<<<2, 128, 0, stream>>>(sum1, sq1, 1.0f / (float)E,
                                         bn1_g + l * 256, bn1_b + l * 256, sc1, sh1, 256);

    edge_pass<0><<<gridE, 128, 0, stream>>>(g, ei, E, sc1, sh1,
                                            w1b + (size_t)l * 128 * 256,
                                            nullptr, nullptr, nullptr,
                                            sum2, sq2, nullptr);
    finalize_bn_k<<<1, 128, 0, stream>>>(sum2, sq2, 1.0f / (float)E,
                                         bn2_g + l * 128, bn2_b + l * 128, sc2, sh2, 128);

    edge_pass<1><<<gridE, 128, 0, stream>>>(g, ei, E, sc1, sh1,
                                            w1b + (size_t)l * 128 * 256,
                                            sc2, sh2,
                                            w2b + (size_t)l * 128 * 128,
                                            nullptr, nullptr, nmax);

    node_finalize_k<<<gridElems, 256, 0, stream>>>(nmax, g, h, N * 128, l > 0 ? 1 : 0);

    if (l < 3) {
      node_stats_k<<<(int)((N + 255) / 256), 128, 0, stream>>>(h, (int)N, sumn, sqn);
      finalize_bn_k<<<1, 128, 0, stream>>>(sumn, sqn, 1.0f / (float)N,
                                           bn_g + l * 128, bn_b + l * 128, scn, shn, 128);
      apply_bn_relu_k<<<gridElems, 256, 0, stream>>>(h, scn, shn, g, N * 128);
    }
  }

  final_linear_k<<<(int)((N + 127) / 128), 128, 0, stream>>>(h, fc_w, fc_b, out, (int)N);
}
